// UltraOptimizedAttention_13932873908518
// MI455X (gfx1250) — matrix operand order repaired and checked by reading, not compiled, not run on hardware
//
#include <hip/hip_runtime.h>
#include <hip/hip_bf16.h>
#include <stdint.h>

// ---------------------------------------------------------------------------
// Problem constants (match reference)
// ---------------------------------------------------------------------------
#define B_   2
#define S_   2048
#define E_   1024
#define H_   16
#define HD_  64
#define M_   (B_ * S_)          // 4096 rows for all projections
#define SCALE_ 0.125f           // 1/sqrt(64)

typedef __attribute__((ext_vector_type(16))) _Float16 v16h;
typedef __attribute__((ext_vector_type(8)))  _Float16 v8h;
typedef __attribute__((ext_vector_type(4)))  _Float16 v4h;
typedef __attribute__((ext_vector_type(8)))  float    v8f;

// ---------------------------------------------------------------------------
// CDNA5 async global->LDS copy (ASYNCcnt-tracked), via inline asm:
//   global_load_async_to_lds_b128 vdst(LDS byte addr), vaddr(64b global), off
// Low 32 bits of a generic LDS pointer are the LDS address (flat aperture).
// ---------------------------------------------------------------------------
static __device__ inline void async_copy_b128(const void* gsrc, void* ldst) {
  const uint64_t ga = (uint64_t)(uintptr_t)gsrc;
  const uint32_t la = (uint32_t)(uintptr_t)ldst;
  asm volatile("global_load_async_to_lds_b128 %0, %1, off"
               :: "v"(la), "v"(ga) : "memory");
}
static __device__ inline void wait_asynccnt_le2() {
  asm volatile("s_wait_asynccnt 0x2" ::: "memory");
}
static __device__ inline void wait_asynccnt_0() {
  asm volatile("s_wait_asynccnt 0x0" ::: "memory");
}

// ---------------------------------------------------------------------------
// WMMA wrapper: D = A(16x32 f16) * B(32x16 f16) + C(16x16 f32)
// ---------------------------------------------------------------------------
static __device__ inline v8f wmma_f16(v16h a, v16h b, v8f c) {
  return __builtin_amdgcn_wmma_f32_16x16x32_f16(
      /*neg_a=*/false, a, /*neg_b=*/false, b,
      /*c_mod=*/(short)0, c, /*reuse_a=*/false, /*reuse_b=*/false);
}

// A-fragment (16x32, 16-bit). lane<16 holds M=lane, K=0..15;
// lane>=16 holds M=lane-16, K=16..31. Two aligned 16B loads.
static __device__ inline v16h load_a_frag(const _Float16* __restrict__ p,
                                          int ld, int m0, int k0, int lane) {
  const int hi = (lane >> 4) & 1;
  const int m  = lane & 15;
  const _Float16* row = p + (size_t)(m0 + m) * ld + k0 + hi * 16;
  const v8h lo = *(const v8h*)(row);        // K j .. j+7
  const v8h hh = *(const v8h*)(row + 8);    // K j+8 .. j+15
  return __builtin_shufflevector(lo, hh, 0, 1, 2, 3, 4, 5, 6, 7,
                                          8, 9, 10, 11, 12, 13, 14, 15);
}

// B-fragment (32x16, 16-bit) from a row-major [N, K] matrix (B[k][n] = M[n][k]):
// lane gives N = n0 + (lane&15); 16 contiguous K per lane (one aligned 32B load).
static __device__ inline v16h load_b_nk(const _Float16* __restrict__ p,
                                        int ld, int n0, int k0, int lane) {
  const int hi = (lane >> 4) & 1;
  const int n  = n0 + (lane & 15);
  const _Float16* src = p + (size_t)n * ld + k0 + hi * 16;
  return *(const v16h*)src;   // 32B-aligned by construction -> 2x b128
}

// B-fragment from an LDS tile stored [64 rows][LDSPITCH halves], row = N.
#define LDSPITCH 40   // 32 data halves + 8 pad halves: 16 distinct banks/frag
static __device__ inline v16h load_b_lds(const _Float16* lp, int t,
                                         int col, int hi) {
  const _Float16* src = lp + (t * 16 + col) * LDSPITCH + hi * 16;
  const v8h lo = *(const v8h*)src;          // 16B-aligned (80B pitch, 32B hi)
  const v8h hh = *(const v8h*)(src + 8);
  return __builtin_shufflevector(lo, hh, 0, 1, 2, 3, 4, 5, 6, 7,
                                          8, 9, 10, 11, 12, 13, 14, 15);
}

// ---------------------------------------------------------------------------
// fp32 -> f16 cast (vectorized, n multiple of 4 — always true here)
// ---------------------------------------------------------------------------
__global__ void cast_f32_to_f16(const float* __restrict__ in,
                                _Float16* __restrict__ out, int n4) {
  const int i = blockIdx.x * blockDim.x + threadIdx.x;
  if (i < n4) {
    const float4 f = ((const float4*)in)[i];
    v4h h;
    h[0] = (_Float16)f.x; h[1] = (_Float16)f.y;
    h[2] = (_Float16)f.z; h[3] = (_Float16)f.w;
    ((v4h*)out)[i] = h;
  }
}

// ---------------------------------------------------------------------------
// GEMM: Y[m,n] = sum_k X[m,k] * W[n,k] + bias[n]     (torch Linear)
// Block = 128 threads (4 waves); block tile 128(M) x 64(N); wave tile 32x64.
// B tile (64x32 f16) staged in LDS via async global->LDS, double-buffered:
// prefetch K-step k+1 while computing k. Per wave: 2 async b128 issues/tile,
// so s_wait_asynccnt 2 after issuing the prefetch => previous tile landed
// (async loads complete in order); __syncthreads() makes it block-visible.
// OUT_MODE 0: f16 [B,H,S,HD]   OUT_MODE 1: f32 [M,N]   OUT_MODE 2: f16 [B,H,HD,S]
// ---------------------------------------------------------------------------
template <int OUT_MODE>
__global__ void __launch_bounds__(128)
gemm_proj(const _Float16* __restrict__ X, const _Float16* __restrict__ W,
          const float* __restrict__ bias, void* __restrict__ outp,
          int Mdim, int Ndim, int Kdim) {
  __shared__ _Float16 bsh[2][64 * LDSPITCH];

  const int tid  = threadIdx.x;
  const int lane = tid & 31;
  const int wave = tid >> 5;
  const int m0 = blockIdx.y * 128 + wave * 32;
  const int n0 = blockIdx.x * 64;

  // Issue the 64x32 B tile for K-step k0 into bsh[buf].
  // 256 chunks of 16B; thread handles chunks 2*tid, 2*tid+1.
  auto issueB = [&](int k0, int buf) {
#pragma unroll
    for (int u = 0; u < 2; ++u) {
      const int c  = tid * 2 + u;
      const int n  = c >> 2;            // 4 chunks per 32-half row
      const int kk = (c & 3) * 8;       // half offset in row
      const _Float16* g = W + (size_t)(n0 + n) * Kdim + k0 + kk;
      _Float16* l = &bsh[buf][n * LDSPITCH + kk];
      async_copy_b128(g, l);
    }
  };

  const int hi  = (lane >> 4) & 1;
  const int col = lane & 15;

  v8f acc[2][4] = {};
  issueB(0, 0);
  for (int k0 = 0; k0 < Kdim; k0 += 32) {
    const int buf = (k0 >> 5) & 1;
    if (k0 + 32 < Kdim) {
      issueB(k0 + 32, buf ^ 1);
      wait_asynccnt_le2();               // tile k0 complete (in-order)
    } else {
      wait_asynccnt_0();
    }
    __syncthreads();                     // all waves' tile-k0 writes visible

    const v16h a0 = load_a_frag(X, Kdim, m0,      k0, lane);
    const v16h a1 = load_a_frag(X, Kdim, m0 + 16, k0, lane);
    const _Float16* lp = bsh[buf];
#pragma unroll
    for (int t = 0; t < 4; ++t) {
      const v16h b = load_b_lds(lp, t, col, hi);
      acc[0][t] = wmma_f16(a0, b, acc[0][t]);
      acc[1][t] = wmma_f16(a1, b, acc[1][t]);
    }
    __syncthreads();                     // done reading before overwrite
  }

#pragma unroll
  for (int i = 0; i < 2; ++i) {
#pragma unroll
    for (int t = 0; t < 4; ++t) {
      const int n = n0 + t * 16 + col;
      const float bv = bias[n];
#pragma unroll
      for (int r = 0; r < 8; ++r) {
        const int m = m0 + i * 16 + r + hi * 8;
        const float val = acc[i][t][r] + bv;
        if (OUT_MODE == 1) {
          ((float*)outp)[(size_t)m * Ndim + n] = val;
        } else {
          const int b = m >> 11;          // m / S_
          const int s = m & (S_ - 1);
          const int h = n >> 6;           // n / HD_
          const int d = n & (HD_ - 1);
          if (OUT_MODE == 0) {            // [B,H,S,HD]
            ((_Float16*)outp)[((((size_t)b * H_ + h) * S_ + s) << 6) + d] =
                (_Float16)val;
          } else {                        // OUT_MODE == 2: [B,H,HD,S] (V^T)
            ((_Float16*)outp)[(((size_t)b * H_ + h) * HD_ + d) * S_ + s] =
                (_Float16)val;
          }
        }
      }
    }
  }
}

// ---------------------------------------------------------------------------
// Flash attention with causal mask and online softmax.
// Grid: (S/64, B*H). Block 128 = 4 waves; wave owns 16 query rows.
// Q/K layout: [B*H, S, HD] f16.  V layout: [B*H, HD, S] f16 (transposed).
// Output: [B, S, E] f16 (heads re-interleaved).
// ---------------------------------------------------------------------------
__global__ void __launch_bounds__(128)
flash_attn(const _Float16* __restrict__ Qp, const _Float16* __restrict__ Kp,
           const _Float16* __restrict__ Vt, _Float16* __restrict__ Oout) {
  // Per-wave P tile, COLUMN-major: element P[row][colk] at lp[colk*16 + row].
  __shared__ _Float16 lds_p[4][32 * 16];

  const int lane = threadIdx.x & 31;
  const int wave = threadIdx.x >> 5;
  const int hi   = (lane >> 4) & 1;
  const int col  = lane & 15;
  const int bh   = blockIdx.y;
  const size_t base = (size_t)bh * S_ * HD_;   // same total for Q/K and V^T
  const int q0 = blockIdx.x * 64 + wave * 16;

  // Preload Q fragments once (HD=64 -> two K=32 chunks)
  const v16h qa0 = load_a_frag(Qp + base, HD_, q0, 0, lane);
  const v16h qa1 = load_a_frag(Qp + base, HD_, q0, 32, lane);

  v8f o[4] = {};
  float mrow[8], lrow[8];
#pragma unroll
  for (int r = 0; r < 8; ++r) { mrow[r] = -3.0e38f; lrow[r] = 0.0f; }

  const int tlast = (q0 + 15) >> 5;   // last 32-wide kv tile touching this wave
  for (int t = 0; t <= tlast; ++t) {
    const int s0 = t * 32;

    // S = Q * K^T : two 16x16 score tiles, accumulated over HD in 2 steps
    v8f sc[2];
#pragma unroll
    for (int nt = 0; nt < 2; ++nt) {
      v8f z = {};
      const v16h bk0 = load_b_nk(Kp + base, HD_, s0 + nt * 16, 0, lane);
      const v8f tmp  = wmma_f16(qa0, bk0, z);
      const v16h bk1 = load_b_nk(Kp + base, HD_, s0 + nt * 16, 32, lane);
      sc[nt] = wmma_f16(qa1, bk1, tmp);
    }

    // Scale + causal mask + row max (half-wave shuffle reduction)
    float p0v[8], p1v[8], tmax[8];
#pragma unroll
    for (int r = 0; r < 8; ++r) {
      const int row = q0 + r + hi * 8;
      float v0 = sc[0][r] * SCALE_;
      float v1 = sc[1][r] * SCALE_;
      if (s0 + col > row)      v0 = -3.0e38f;
      if (s0 + 16 + col > row) v1 = -3.0e38f;
      p0v[r] = v0; p1v[r] = v1;
      float mx = fmaxf(v0, v1);
#pragma unroll
      for (int xm = 1; xm < 16; xm <<= 1) mx = fmaxf(mx, __shfl_xor(mx, xm, 32));
      tmax[r] = mx;
    }

    // Online softmax update
#pragma unroll
    for (int r = 0; r < 8; ++r) {
      const float mnew  = fmaxf(mrow[r], tmax[r]);
      const float alpha = __expf(mrow[r] - mnew);
      const float e0 = __expf(p0v[r] - mnew);
      const float e1 = __expf(p1v[r] - mnew);
      p0v[r] = e0; p1v[r] = e1;
      float rs = e0 + e1;
#pragma unroll
      for (int xm = 1; xm < 16; xm <<= 1) rs += __shfl_xor(rs, xm, 32);
      lrow[r] = lrow[r] * alpha + rs;
      mrow[r] = mnew;
#pragma unroll
      for (int dt = 0; dt < 4; ++dt) o[dt][r] *= alpha;
    }

    // P (C-layout f32) -> LDS f16 (column-major; two b128 stores per lane)
    _Float16* lp = lds_p[wave];
    {
      v8h ph0, ph1;
#pragma unroll
      for (int r = 0; r < 8; ++r) {
        ph0[r] = (_Float16)p0v[r];
        ph1[r] = (_Float16)p1v[r];
      }
      *(v8h*)(lp + (col)      * 16 + hi * 8) = ph0;  // tile 0, column col
      *(v8h*)(lp + (16 + col) * 16 + hi * 8) = ph1;  // tile 1, column col+16
    }
    // Read back as A-fragment (transpose side: strided scalar ds loads)
    v16h pa;
#pragma unroll
    for (int j = 0; j < 16; ++j) {
      const int k = j + hi * 16;
      pa[j] = lp[k * 16 + col];
    }

    // O += P * V   (V^T is [HD, S] row-major -> contiguous B-fragments)
#pragma unroll
    for (int dt = 0; dt < 4; ++dt) {
      const v16h bv = load_b_nk(Vt + base, S_, dt * 16, s0, lane);
      o[dt] = wmma_f16(pa, bv, o[dt]);
    }
  }

  // Normalize and write [B, S, E] f16 with heads re-interleaved
  const int b = bh / H_;
  const int h = bh % H_;
#pragma unroll
  for (int r = 0; r < 8; ++r) {
    const float inv = 1.0f / lrow[r];
    const int q = q0 + r + hi * 8;
    const size_t rowbase = ((size_t)b * S_ + q) * E_ + h * HD_;
#pragma unroll
    for (int dt = 0; dt < 4; ++dt) {
      Oout[rowbase + dt * 16 + col] = (_Float16)(o[dt][r] * inv);
    }
  }
}

// ---------------------------------------------------------------------------
// Launcher
// ---------------------------------------------------------------------------
extern "C" void kernel_launch(void* const* d_in, const int* in_sizes, int n_in,
                              void* d_out, int out_size, void* d_ws, size_t ws_size,
                              hipStream_t stream) {
  const float* q  = (const float*)d_in[0];
  const float* k  = (const float*)d_in[1];
  const float* v  = (const float*)d_in[2];
  const float* Wq = (const float*)d_in[3];
  const float* Wk = (const float*)d_in[4];
  const float* Wv = (const float*)d_in[5];
  const float* Wo = (const float*)d_in[6];
  const float* bq = (const float*)d_in[7];
  const float* bk = (const float*)d_in[8];
  const float* bv = (const float*)d_in[9];
  const float* bo = (const float*)d_in[10];

  const size_t ACT = (size_t)M_ * E_;   // 4,194,304 elems
  const size_t WEL = (size_t)E_ * E_;   // 1,048,576 elems

  char* ws = (char*)d_ws;
  size_t off = 0;
  auto carve = [&](size_t bytes) -> void* {
    void* p = ws + off;
    off += (bytes + 255) & ~(size_t)255;
    return p;
  };

  _Float16* q16  = (_Float16*)carve(ACT * 2);   // reused later as attn output
  _Float16* k16  = (_Float16*)carve(ACT * 2);
  _Float16* v16  = (_Float16*)carve(ACT * 2);
  _Float16* wq16 = (_Float16*)carve(WEL * 2);
  _Float16* wk16 = (_Float16*)carve(WEL * 2);
  _Float16* wv16 = (_Float16*)carve(WEL * 2);
  _Float16* wo16 = (_Float16*)carve(WEL * 2);
  _Float16* Qp   = (_Float16*)carve(ACT * 2);   // [B*H, S, HD]
  _Float16* Kp   = (_Float16*)carve(ACT * 2);   // [B*H, S, HD]
  _Float16* Vt   = (_Float16*)carve(ACT * 2);   // [B*H, HD, S]  (V transposed)
  _Float16* attn16 = q16;  // q16 dead after Q projection; reuse for attn output

  const int CT = 256;
  const int actB = (int)((ACT / 4 + CT - 1) / CT);
  const int welB = (int)((WEL / 4 + CT - 1) / CT);
  cast_f32_to_f16<<<actB, CT, 0, stream>>>(q,  q16,  (int)(ACT / 4));
  cast_f32_to_f16<<<actB, CT, 0, stream>>>(k,  k16,  (int)(ACT / 4));
  cast_f32_to_f16<<<actB, CT, 0, stream>>>(v,  v16,  (int)(ACT / 4));
  cast_f32_to_f16<<<welB, CT, 0, stream>>>(Wq, wq16, (int)(WEL / 4));
  cast_f32_to_f16<<<welB, CT, 0, stream>>>(Wk, wk16, (int)(WEL / 4));
  cast_f32_to_f16<<<welB, CT, 0, stream>>>(Wv, wv16, (int)(WEL / 4));
  cast_f32_to_f16<<<welB, CT, 0, stream>>>(Wo, wo16, (int)(WEL / 4));

  const dim3 gGemm(E_ / 64, M_ / 128);   // (16, 32)
  gemm_proj<0><<<gGemm, 128, 0, stream>>>(q16, wq16, bq, Qp, M_, E_, E_);
  gemm_proj<0><<<gGemm, 128, 0, stream>>>(k16, wk16, bk, Kp, M_, E_, E_);
  gemm_proj<2><<<gGemm, 128, 0, stream>>>(v16, wv16, bv, Vt, M_, E_, E_);

  const dim3 gAttn(S_ / 64, B_ * H_);    // (32, 32)
  flash_attn<<<gAttn, 128, 0, stream>>>(Qp, Kp, Vt, attn16);

  gemm_proj<1><<<gGemm, 128, 0, stream>>>(attn16, wo16, bo, d_out, M_, E_, E_);

  (void)in_sizes; (void)n_in; (void)out_size; (void)ws_size;
}
